// GRUCell_50165218017581
// MI455X (gfx1250) — compile-verified, matrix operand order checked by reading, not run on hardware
//
#include <hip/hip_runtime.h>
#include <hip/hip_bf16.h>

typedef __bf16 bf16;
typedef __attribute__((ext_vector_type(16))) __bf16 v16bf;
typedef __attribute__((ext_vector_type(8)))  __bf16 v8bf;
typedef __attribute__((ext_vector_type(8)))  float  v8f;

// Explicit global address space so GEMM-loop loads lower to global_load_b128
// (LOADcnt only), never flat_load_b128 (LOADcnt+DScnt, LDS-path arbitration).
#define GAS __attribute__((address_space(1)))
typedef GAS const bf16* gcptr;

#define B_ROWS 16384
#define DIM_D  512
#define DIM_U  512
#define DIM_K  1024   // D + U

// Fragment gather for v_wmma_f32_16x16x32_bf16 (A layout; B identical on W^T rows).
// Lane half h in {0,1}: elems 0..7 = K[h*8 .. +7], elems 8..15 = K[16 + h*8 .. +7].
// p is the per-lane base (row + h*8 applied); ko is a compile-time element offset.
__device__ __forceinline__ v16bf frag_at(gcptr p, int ko) {
    v8bf lo = *(const GAS v8bf*)(p + ko);
    v8bf hi = *(const GAS v8bf*)(p + ko + 16);
    return __builtin_shufflevector(lo, hi, 0,1,2,3,4,5,6,7,8,9,10,11,12,13,14,15);
}

__device__ __forceinline__ float fast_sigmoid(float x) {
    x = fminf(fmaxf(x, -30.f), 30.f);
    float e = __expf(-x);
    return __fdividef(1.f, 1.f + e);
}

__device__ __forceinline__ float fast_tanh(float x) {
    x = fminf(fmaxf(x, -15.f), 15.f);
    float e = __expf(2.f * x);
    return __fdividef(e - 1.f, e + 1.f);
}

// One K-chunk of 64: two sub-steps with immediate offsets, then bump pointers.
template <int NB>
__device__ __forceinline__ void mma_chunk64(gcptr (&ap)[4], gcptr (&bp)[NB],
                                            v8f (&acc)[4][NB]) {
    #pragma unroll
    for (int u = 0; u < 2; ++u) {
        const int ko = u * 32;
        v16bf a[4], b[NB];
        #pragma unroll
        for (int mt = 0; mt < 4; ++mt) a[mt] = frag_at(ap[mt], ko);
        #pragma unroll
        for (int j = 0; j < NB; ++j)   b[j]  = frag_at(bp[j], ko);
        #pragma unroll
        for (int mt = 0; mt < 4; ++mt) {
            #pragma unroll
            for (int j = 0; j < NB; ++j) {
                acc[mt][j] = __builtin_amdgcn_wmma_f32_16x16x32_bf16(
                    false, a[mt], false, b[j], (short)0, acc[mt][j], false, false);
            }
        }
    }
    #pragma unroll
    for (int mt = 0; mt < 4; ++mt) ap[mt] += 64;
    #pragma unroll
    for (int j = 0; j < NB; ++j)   bp[j]  += 64;
}

// ---- Phase 0a: fusedA = bf16([inputs | h_prev]), [B, 1024]; 8 elems/thread ----
__global__ __launch_bounds__(256) void fuse_convert_kernel(const float* __restrict__ x,
                                                           const float* __restrict__ h,
                                                           bf16* __restrict__ fusedA) {
    size_t base = ((size_t)blockIdx.x * 256 + threadIdx.x) * 8;
    int    col  = (int)(base & 1023);
    size_t row  = base >> 10;
    const float* src = (col < DIM_D) ? (x + row * DIM_D + col)
                                     : (h + row * DIM_U + (col - DIM_D));
    v8bf o;
    #pragma unroll
    for (int i = 0; i < 8; ++i) o[i] = (bf16)src[i];
    *(v8bf*)(fusedA + base) = o;
}

// ---- Phase 0b: WT[n,k] = bf16(W[k,n]); 8 k's/thread (coalesced 16B writes) ----
__global__ __launch_bounds__(256) void transpose_w_kernel(const float* __restrict__ W,
                                                          bf16* __restrict__ WT) {
    size_t base = ((size_t)blockIdx.x * 256 + threadIdx.x) * 8;  // n*1024 + k
    int k = (int)(base & 1023);
    int n = (int)(base >> 10);
    v8bf o;
    #pragma unroll
    for (int i = 0; i < 8; ++i) o[i] = (bf16)W[(size_t)(k + i) * DIM_U + n];
    *(v8bf*)(WT + base) = o;
}

// ---- Phase 1: Z = sigmoid(fusedA@Wz+bz), R = sigmoid(fusedA@Wr+br);
//      write zg (f32) and A2h = bf16(R * h_prev). ----
// Waves arranged 4(M) x 2(N): block tile 256(M) x 64(N), both gates.
// Intra-block A-dup (2x) and B-dup (4x) absorbed by WGP$; L2 reads minimized.
// Grid: (512/64, 16384/256) = (8, 64).
__global__ __launch_bounds__(256) void gate_gemm_kernel(
    const bf16* __restrict__ fusedA, const bf16* __restrict__ WzT,
    const bf16* __restrict__ WrT,
    const float* __restrict__ bz, const float* __restrict__ br,
    const float* __restrict__ h_prev,
    float* __restrict__ zg, bf16* __restrict__ A2h)
{
    const int lane = threadIdx.x & 31;
    const int wave = threadIdx.x >> 5;
    const int m0 = blockIdx.y * 256 + (wave >> 1) * 64;
    const int n0 = blockIdx.x * 64  + (wave & 1) * 32;
    const int lrow = lane & 15;
    const int lh   = (lane >> 4) * 8;

    gcptr ap[4];
    gcptr bp[4];   // [0..1] = Wz tiles, [2..3] = Wr tiles
    #pragma unroll
    for (int mt = 0; mt < 4; ++mt)
        ap[mt] = (gcptr)fusedA + (size_t)(m0 + mt * 16 + lrow) * DIM_K + lh;
    #pragma unroll
    for (int nt = 0; nt < 2; ++nt) {
        bp[nt]     = (gcptr)WzT + (size_t)(n0 + nt * 16 + lrow) * DIM_K + lh;
        bp[2 + nt] = (gcptr)WrT + (size_t)(n0 + nt * 16 + lrow) * DIM_K + lh;
    }

    v8f acc[4][4] = {};   // [mt][0..1]=Z, [mt][2..3]=R

    for (int k = 0; k < DIM_K; k += 64)
        mma_chunk64<4>(ap, bp, acc);

    // C/D layout: element r of lane L -> m = (L>>4)*8 + r, n = L&15
    const int mh = (lane >> 4) * 8;
    #pragma unroll
    for (int mt = 0; mt < 4; ++mt) {
        #pragma unroll
        for (int nt = 0; nt < 2; ++nt) {
            const int col = n0 + nt * 16 + lrow;
            const float bzv = bz[col];
            const float brv = br[col];
            #pragma unroll
            for (int r = 0; r < 8; ++r) {
                const int row = m0 + mt * 16 + mh + r;
                const size_t o = (size_t)row * DIM_U + col;
                const float z  = fast_sigmoid(acc[mt][nt][r] + bzv);
                const float rr = fast_sigmoid(acc[mt][2 + nt][r] + brv);
                zg[o]  = z;
                A2h[o] = (bf16)(rr * h_prev[o]);
            }
        }
    }
}

// ---- Phase 2: h_tilde = tanh([inputs | rg*h_prev] @ Wh + bh);
//      out = h_prev + z*(h_tilde - h_prev). ----
// Waves 4(M) x 2(N): block tile 256(M) x 128(N). Grid: (4, 64).
__global__ __launch_bounds__(256) void hidden_gemm_kernel(
    const bf16* __restrict__ fusedA, const bf16* __restrict__ A2h,
    const bf16* __restrict__ WhT, const float* __restrict__ bh,
    const float* __restrict__ zg, const float* __restrict__ h_prev,
    float* __restrict__ out)
{
    const int lane = threadIdx.x & 31;
    const int wave = threadIdx.x >> 5;
    const int m0 = blockIdx.y * 256 + (wave >> 1) * 64;
    const int n0 = blockIdx.x * 128 + (wave & 1) * 64;
    const int lrow = lane & 15;
    const int lh   = (lane >> 4) * 8;

    gcptr ap[4];
    gcptr bp[4];
    #pragma unroll
    for (int mt = 0; mt < 4; ++mt)
        ap[mt] = (gcptr)fusedA + (size_t)(m0 + mt * 16 + lrow) * DIM_K + lh;
    #pragma unroll
    for (int nt = 0; nt < 4; ++nt)
        bp[nt] = (gcptr)WhT + (size_t)(n0 + nt * 16 + lrow) * DIM_K + lh;

    v8f acc[4][4] = {};

    // First half of K: inputs columns of fusedA
    for (int k = 0; k < DIM_D; k += 64)
        mma_chunk64<4>(ap, bp, acc);

    // Second half of K: rg * h_prev from A2h (B pointers keep advancing)
    #pragma unroll
    for (int mt = 0; mt < 4; ++mt)
        ap[mt] = (gcptr)A2h + (size_t)(m0 + mt * 16 + lrow) * DIM_U + lh;
    for (int k = 0; k < DIM_U; k += 64)
        mma_chunk64<4>(ap, bp, acc);

    const int mh = (lane >> 4) * 8;
    #pragma unroll
    for (int mt = 0; mt < 4; ++mt) {
        #pragma unroll
        for (int nt = 0; nt < 4; ++nt) {
            const int col = n0 + nt * 16 + lrow;
            const float bhv = bh[col];
            #pragma unroll
            for (int r = 0; r < 8; ++r) {
                const int row = m0 + mt * 16 + mh + r;
                const size_t o = (size_t)row * DIM_U + col;
                const float ht = fast_tanh(acc[mt][nt][r] + bhv);
                const float z  = zg[o];
                const float h  = h_prev[o];
                out[o] = fmaf(z, ht - h, h);
            }
        }
    }
}

extern "C" void kernel_launch(void* const* d_in, const int* in_sizes, int n_in,
                              void* d_out, int out_size, void* d_ws, size_t ws_size,
                              hipStream_t stream) {
    const float* inputs = (const float*)d_in[0];
    const float* h_prev = (const float*)d_in[1];
    const float* Wz     = (const float*)d_in[2];
    const float* Wr     = (const float*)d_in[3];
    const float* Wh     = (const float*)d_in[4];
    const float* bz     = (const float*)d_in[5];
    const float* br     = (const float*)d_in[6];
    const float* bh     = (const float*)d_in[7];
    float* out = (float*)d_out;

    char* ws = (char*)d_ws;
    bf16* fusedA = (bf16*)ws;  ws += (size_t)B_ROWS * DIM_K * sizeof(bf16);   // 32 MB
    bf16* WzT    = (bf16*)ws;  ws += (size_t)DIM_U  * DIM_K * sizeof(bf16);   //  1 MB
    bf16* WrT    = (bf16*)ws;  ws += (size_t)DIM_U  * DIM_K * sizeof(bf16);   //  1 MB
    bf16* WhT    = (bf16*)ws;  ws += (size_t)DIM_U  * DIM_K * sizeof(bf16);   //  1 MB
    bf16* A2h    = (bf16*)ws;  ws += (size_t)B_ROWS * DIM_U * sizeof(bf16);   // 16 MB
    float* zgbuf = (float*)ws;                                                // 32 MB

    fuse_convert_kernel<<<(B_ROWS * DIM_K) / (256 * 8), 256, 0, stream>>>(inputs, h_prev, fusedA);
    transpose_w_kernel<<<(DIM_U * DIM_K) / (256 * 8), 256, 0, stream>>>(Wz, WzT);
    transpose_w_kernel<<<(DIM_U * DIM_K) / (256 * 8), 256, 0, stream>>>(Wr, WrT);
    transpose_w_kernel<<<(DIM_U * DIM_K) / (256 * 8), 256, 0, stream>>>(Wh, WhT);

    gate_gemm_kernel<<<dim3(DIM_U / 64, B_ROWS / 256), 256, 0, stream>>>(
        fusedA, WzT, WrT, bz, br, h_prev, zgbuf, A2h);

    hidden_gemm_kernel<<<dim3(DIM_U / 128, B_ROWS / 256), 256, 0, stream>>>(
        fusedA, A2h, WhT, bh, zgbuf, h_prev, out);
}
